// BatchTopKSAE_24111946400110
// MI455X (gfx1250) — matrix operand order repaired and checked by reading, not compiled, not run on hardware
//
#include <hip/hip_runtime.h>
#include <hip/hip_bf16.h>

// ---------------------------------------------------------------------------
// BatchTopK SAE forward for MI455X (gfx1250), wave32 + WMMA bf16 path.
//   1) prep_x: xc_bf16 = bf16(x - b_dec)
//   2) transpose_convert: W_enc -> WencT bf16 [H][K], W_dec -> WdecT bf16 [D][H]
//   3) gemm_enc: LDS double-buffered, v_wmma_f32_16x16x32_bf16 -> pre_acts f32
//   4) two-level histogram radix select -> global top-k threshold
//   5) scatter_topk -> acts_topk bf16 (+ exact l0 count, integer atomics)
//   6) aux_topk: per-row top-64 over dead cols -> acts_aux bf16
//   7) gemm_dec fused (topk+aux share W_dec B fragments via LDS):
//        x_rec -> d_out, per-block deterministic l2/aux partial sums
//   8) finalize: fixed-order reduction -> loss, l2, l0, aux scalars
// ---------------------------------------------------------------------------

#define BATCH   2048
#define IN_DIM  1024
#define HID_DIM 16384
#define TOPK    128
#define TOPK_AUX 64
#define N_DEAD  100
#define AUX_PEN 0.01f

#define KSTEP 32
#define LPITCH 40   // bf16 elements per LDS row (80B): conflict-free b128 frags

typedef __attribute__((ext_vector_type(16))) __bf16 v16bf;
typedef __attribute__((ext_vector_type(8)))  float  v8f;
typedef __attribute__((ext_vector_type(4)))  unsigned int u32x4;

static_assert(sizeof(v16bf) == 32, "v16bf must be 32 bytes");

union FragAB {
    v16bf v;
    u32x4 q[2];
};

// ------------------------- conversion kernels ------------------------------

__global__ __launch_bounds__(256) void prep_x_kernel(const float* __restrict__ x,
                                                     const float* __restrict__ b_dec,
                                                     __bf16* __restrict__ xc) {
    const size_t total = (size_t)BATCH * IN_DIM;
    for (size_t i = (size_t)blockIdx.x * blockDim.x + threadIdx.x; i < total;
         i += (size_t)gridDim.x * blockDim.x) {
        xc[i] = (__bf16)(x[i] - b_dec[i & (IN_DIM - 1)]);
    }
}

// in: f32 [K][N] row-major, out: bf16 [N][K] row-major (K-major per column)
__global__ __launch_bounds__(256) void transpose_convert_kernel(const float* __restrict__ in,
                                                                __bf16* __restrict__ out,
                                                                int K, int N) {
    __shared__ float tile[32][33];
    const int k0 = blockIdx.y * 32;
    const int n0 = blockIdx.x * 32;
    const int tx = threadIdx.x & 31;
    const int ty = threadIdx.x >> 5;  // 0..7
    #pragma unroll
    for (int r = ty; r < 32; r += 8)
        tile[r][tx] = in[(size_t)(k0 + r) * N + (n0 + tx)];
    __syncthreads();
    #pragma unroll
    for (int r = ty; r < 32; r += 8)
        out[(size_t)(n0 + r) * K + (k0 + tx)] = (__bf16)tile[tx][r];
}

// ------------------------- encoder GEMM (WMMA) -----------------------------
// pre[m][n] = sum_k xc[m][k] * WencT[n][k] ;  M=2048 N=16384 K=1024
// Block tile 64(M) x 128(N), 8 waves (4x2), wave tile 16x64.
// LDS double-buffered staging: one __syncthreads per k-step; staged global
// loads for step k+1 are issued before the WMMAs of step k.

__global__ __launch_bounds__(256) void gemm_enc_kernel(const __bf16* __restrict__ Xc,
                                                       const __bf16* __restrict__ WencT,
                                                       float* __restrict__ P) {
    const int K = IN_DIM, N = HID_DIM;
    __shared__ __bf16 sA[2][64 * LPITCH];    // 10 KB
    __shared__ __bf16 sB[2][128 * LPITCH];   // 20 KB

    const int tid  = threadIdx.x;
    const int wid  = tid >> 5;
    const int lane = tid & 31;
    const int half = lane >> 4;
    const int l15  = lane & 15;
    const int m0b = blockIdx.y * 64;
    const int n0b = blockIdx.x * 128;
    const int wm = wid >> 1;   // 0..3
    const int wn = wid & 1;    // 0..1

    // staging indices (per thread): A one b128, B two b128 per k-step
    const int ar = tid >> 2;            // 0..63
    const int ac = (tid & 3) * 8;       // 0,8,16,24
    const int bn0 = tid >> 2;           // 0..63
    const int bn1 = (tid + 256) >> 2;   // 64..127
    const int bc  = (tid & 3) * 8;

    const __bf16* aGsrc = Xc + (size_t)(m0b + ar) * K + ac;
    const __bf16* bGsrc0 = WencT + (size_t)(n0b + bn0) * K + bc;
    const __bf16* bGsrc1 = WencT + (size_t)(n0b + bn1) * K + bc;

    v8f c0 = {}, c1 = {}, c2 = {}, c3 = {};

    // prologue: stage k=0 into buffer 0
    {
        u32x4 ra  = *(const u32x4*)(aGsrc);
        u32x4 rb0 = *(const u32x4*)(bGsrc0);
        u32x4 rb1 = *(const u32x4*)(bGsrc1);
        *(u32x4*)(&sA[0][ar * LPITCH + ac])   = ra;
        *(u32x4*)(&sB[0][bn0 * LPITCH + bc])  = rb0;
        *(u32x4*)(&sB[0][bn1 * LPITCH + bc])  = rb1;
    }
    __syncthreads();

    int buf = 0;
    for (int k = KSTEP; k <= K; k += KSTEP) {
        u32x4 ra, rb0, rb1;
        const bool more = (k < K);
        if (more) {
            ra  = *(const u32x4*)(aGsrc + k);
            rb0 = *(const u32x4*)(bGsrc0 + k);
            rb1 = *(const u32x4*)(bGsrc1 + k);
        }

        // compute current buffer from LDS
        FragAB a;
        const __bf16* aL = &sA[buf][(wm * 16 + l15) * LPITCH + half * 8];
        a.q[0] = *(const u32x4*)(aL);
        a.q[1] = *(const u32x4*)(aL + 16);
        #pragma unroll
        for (int t = 0; t < 4; ++t) {
            FragAB b;
            const __bf16* bL = &sB[buf][(wn * 64 + t * 16 + l15) * LPITCH + half * 8];
            b.q[0] = *(const u32x4*)(bL);
            b.q[1] = *(const u32x4*)(bL + 16);
            switch (t) {
            case 0: c0 = __builtin_amdgcn_wmma_f32_16x16x32_bf16(false, a.v, false, b.v, (short)0, c0, false, false); break;
            case 1: c1 = __builtin_amdgcn_wmma_f32_16x16x32_bf16(false, a.v, false, b.v, (short)0, c1, false, false); break;
            case 2: c2 = __builtin_amdgcn_wmma_f32_16x16x32_bf16(false, a.v, false, b.v, (short)0, c2, false, false); break;
            default: c3 = __builtin_amdgcn_wmma_f32_16x16x32_bf16(false, a.v, false, b.v, (short)0, c3, false, false); break;
            }
        }

        if (more) {
            const int nb = buf ^ 1;
            *(u32x4*)(&sA[nb][ar * LPITCH + ac])  = ra;
            *(u32x4*)(&sB[nb][bn0 * LPITCH + bc]) = rb0;
            *(u32x4*)(&sB[nb][bn1 * LPITCH + bc]) = rb1;
            __syncthreads();
            buf = nb;
        }
    }

    const int m0 = m0b + wm * 16;
    const int n0 = n0b + wn * 64;
    v8f acc[4] = {c0, c1, c2, c3};
    #pragma unroll
    for (int t = 0; t < 4; ++t) {
        const int col = n0 + t * 16 + l15;
        #pragma unroll
        for (int r = 0; r < 8; ++r) {
            const int m = m0 + r + half * 8;
            P[(size_t)m * N + col] = acc[t][r];
        }
    }
}

// ------------------ global top-k threshold (radix select) ------------------

#define NBINS 8192

__global__ __launch_bounds__(256) void hist1_kernel(const float* __restrict__ P,
                                                    unsigned int* __restrict__ hist) {
    __shared__ unsigned int h[NBINS];
    for (int i = threadIdx.x; i < NBINS; i += 256) h[i] = 0u;
    __syncthreads();
    const size_t total = (size_t)BATCH * HID_DIM;
    for (size_t i = (size_t)blockIdx.x * 256 + threadIdx.x; i < total;
         i += (size_t)gridDim.x * 256) {
        float v = P[i];
        if (v > 0.0f) atomicAdd(&h[(__float_as_uint(v) >> 18) & (NBINS - 1)], 1u);
    }
    __syncthreads();
    for (int i = threadIdx.x; i < NBINS; i += 256)
        if (h[i]) atomicAdd(&hist[i], h[i]);
}

__global__ void scan1_kernel(const unsigned int* __restrict__ hist,
                             unsigned int* __restrict__ ctrl) {
    if (threadIdx.x == 0 && blockIdx.x == 0) {
        const long long need = (long long)TOPK * BATCH;
        long long cum = 0;
        int t1 = -1;
        for (int b = NBINS - 1; b >= 0; --b) {
            long long prev = cum;
            cum += hist[b];
            if (cum >= need) { t1 = b; ctrl[0] = (unsigned)b; ctrl[1] = (unsigned)(need - prev); break; }
        }
        if (t1 < 0) { ctrl[0] = 0u; ctrl[1] = 0xFFFFFFFFu; }
    }
}

__global__ __launch_bounds__(256) void hist2_kernel(const float* __restrict__ P,
                                                    const unsigned int* __restrict__ ctrl,
                                                    unsigned int* __restrict__ hist) {
    __shared__ unsigned int h[NBINS];
    for (int i = threadIdx.x; i < NBINS; i += 256) h[i] = 0u;
    __syncthreads();
    const unsigned t1 = ctrl[0];
    const size_t total = (size_t)BATCH * HID_DIM;
    for (size_t i = (size_t)blockIdx.x * 256 + threadIdx.x; i < total;
         i += (size_t)gridDim.x * 256) {
        float v = P[i];
        if (v > 0.0f) {
            unsigned bits = __float_as_uint(v);
            if (((bits >> 18) & (NBINS - 1)) == t1)
                atomicAdd(&h[(bits >> 5) & (NBINS - 1)], 1u);
        }
    }
    __syncthreads();
    for (int i = threadIdx.x; i < NBINS; i += 256)
        if (h[i]) atomicAdd(&hist[i], h[i]);
}

__global__ void scan2_kernel(const unsigned int* __restrict__ hist2,
                             unsigned int* __restrict__ ctrl) {
    if (threadIdx.x == 0 && blockIdx.x == 0) {
        const unsigned t1 = ctrl[0];
        const unsigned long long need = ctrl[1];
        unsigned long long cum = 0;
        unsigned t2 = 0;
        bool found = false;
        for (int b = NBINS - 1; b >= 0; --b) {
            cum += hist2[b];
            if (cum >= need) { t2 = (unsigned)b; found = true; break; }
        }
        ctrl[2] = found ? ((t1 << 18) | (t2 << 5)) : (t1 << 18);
    }
}

__global__ __launch_bounds__(256) void scatter_topk_kernel(const float* __restrict__ P,
                                                           const unsigned int* __restrict__ ctrl,
                                                           __bf16* __restrict__ Atopk,
                                                           unsigned int* __restrict__ selcount) {
    __shared__ unsigned int cnt[256];
    const unsigned thr = ctrl[2];
    unsigned local = 0;
    const size_t total = (size_t)BATCH * HID_DIM;
    for (size_t i = (size_t)blockIdx.x * 256 + threadIdx.x; i < total;
         i += (size_t)gridDim.x * 256) {
        float v = P[i];
        bool sel = (v > 0.0f) && (__float_as_uint(v) >= thr);
        Atopk[i] = sel ? (__bf16)v : (__bf16)0.0f;
        local += sel ? 1u : 0u;
    }
    cnt[threadIdx.x] = local;
    __syncthreads();
    for (int s = 128; s > 0; s >>= 1) {
        if (threadIdx.x < s) cnt[threadIdx.x] += cnt[threadIdx.x + s];
        __syncthreads();
    }
    if (threadIdx.x == 0 && cnt[0]) atomicAdd(selcount, cnt[0]);
}

// ------------------- per-row aux top-64 over dead columns ------------------

__global__ __launch_bounds__(256) void aux_topk_kernel(const float* __restrict__ P,
                                                       const int* __restrict__ nba,
                                                       __bf16* __restrict__ Aaux) {
    __shared__ unsigned int h[NBINS];
    __shared__ unsigned int sThr;
    const int row = blockIdx.x;
    const float* pr = P + (size_t)row * HID_DIM;
    for (int i = threadIdx.x; i < NBINS; i += 256) h[i] = 0u;
    __syncthreads();
    for (int j = threadIdx.x; j < HID_DIM; j += 256) {
        if (nba[j] >= N_DEAD) {
            float v = pr[j];
            if (v > 0.0f) atomicAdd(&h[(__float_as_uint(v) >> 18) & (NBINS - 1)], 1u);
        }
    }
    __syncthreads();
    if (threadIdx.x == 0) {
        unsigned cum = 0, t = 0;
        for (int b = NBINS - 1; b >= 0; --b) {
            cum += h[b];
            if (cum >= TOPK_AUX) { t = (unsigned)b; break; }
        }
        sThr = (cum >= TOPK_AUX) ? (t << 18) : 0u;  // <64 available -> take all positives
    }
    __syncthreads();
    const unsigned thr = sThr;
    __bf16* outr = Aaux + (size_t)row * HID_DIM;
    for (int j = threadIdx.x; j < HID_DIM; j += 256) {
        float v = 0.0f;
        if (nba[j] >= N_DEAD) {
            float p = pr[j];
            if (p > 0.0f && __float_as_uint(p) >= thr) v = p;
        }
        outr[j] = (__bf16)v;
    }
}

// ---------------- fused decoder GEMM + loss partials (WMMA) ----------------
// x_rec = Atopk @ Wdec + b_dec ; x_aux = Aaux @ Wdec  (shared B fragments)
// M=2048 N=1024 K=16384; LDS double-buffered like gemm_enc.

__global__ __launch_bounds__(256) void gemm_dec_kernel(const __bf16* __restrict__ Atopk,
                                                       const __bf16* __restrict__ Aaux,
                                                       const __bf16* __restrict__ WdecT,
                                                       const float* __restrict__ x,
                                                       const float* __restrict__ b_dec,
                                                       float* __restrict__ out,
                                                       float* __restrict__ partials) {
    const int K = HID_DIM, N = IN_DIM;
    __shared__ __bf16 sAt[2][64 * LPITCH];   // 10 KB
    __shared__ __bf16 sAa[2][64 * LPITCH];   // 10 KB
    __shared__ __bf16 sB[2][128 * LPITCH];   // 20 KB

    const int tid  = threadIdx.x;
    const int wid  = tid >> 5;
    const int lane = tid & 31;
    const int half = lane >> 4;
    const int l15  = lane & 15;
    const int m0b = blockIdx.y * 64;
    const int n0b = blockIdx.x * 128;
    const int wm = wid >> 1;
    const int wn = wid & 1;

    const int ar = tid >> 2;
    const int ac = (tid & 3) * 8;
    const int bn0 = tid >> 2;
    const int bn1 = (tid + 256) >> 2;
    const int bc  = (tid & 3) * 8;

    const __bf16* atG = Atopk + (size_t)(m0b + ar) * K + ac;
    const __bf16* aaG = Aaux  + (size_t)(m0b + ar) * K + ac;
    const __bf16* bG0 = WdecT + (size_t)(n0b + bn0) * K + bc;
    const __bf16* bG1 = WdecT + (size_t)(n0b + bn1) * K + bc;

    v8f ct0 = {}, ct1 = {}, ct2 = {}, ct3 = {};
    v8f ca0 = {}, ca1 = {}, ca2 = {}, ca3 = {};

    {
        u32x4 rt  = *(const u32x4*)(atG);
        u32x4 rq  = *(const u32x4*)(aaG);
        u32x4 rb0 = *(const u32x4*)(bG0);
        u32x4 rb1 = *(const u32x4*)(bG1);
        *(u32x4*)(&sAt[0][ar * LPITCH + ac])  = rt;
        *(u32x4*)(&sAa[0][ar * LPITCH + ac])  = rq;
        *(u32x4*)(&sB[0][bn0 * LPITCH + bc])  = rb0;
        *(u32x4*)(&sB[0][bn1 * LPITCH + bc])  = rb1;
    }
    __syncthreads();

    int buf = 0;
    for (int k = KSTEP; k <= K; k += KSTEP) {
        u32x4 rt, rq, rb0, rb1;
        const bool more = (k < K);
        if (more) {
            rt  = *(const u32x4*)(atG + k);
            rq  = *(const u32x4*)(aaG + k);
            rb0 = *(const u32x4*)(bG0 + k);
            rb1 = *(const u32x4*)(bG1 + k);
        }

        FragAB at, aa;
        const __bf16* atL = &sAt[buf][(wm * 16 + l15) * LPITCH + half * 8];
        const __bf16* aaL = &sAa[buf][(wm * 16 + l15) * LPITCH + half * 8];
        at.q[0] = *(const u32x4*)(atL);
        at.q[1] = *(const u32x4*)(atL + 16);
        aa.q[0] = *(const u32x4*)(aaL);
        aa.q[1] = *(const u32x4*)(aaL + 16);
        #pragma unroll
        for (int t = 0; t < 4; ++t) {
            FragAB b;
            const __bf16* bL = &sB[buf][(wn * 64 + t * 16 + l15) * LPITCH + half * 8];
            b.q[0] = *(const u32x4*)(bL);
            b.q[1] = *(const u32x4*)(bL + 16);
            switch (t) {
            case 0:
                ct0 = __builtin_amdgcn_wmma_f32_16x16x32_bf16(false, at.v, false, b.v, (short)0, ct0, false, false);
                ca0 = __builtin_amdgcn_wmma_f32_16x16x32_bf16(false, aa.v, false, b.v, (short)0, ca0, false, false);
                break;
            case 1:
                ct1 = __builtin_amdgcn_wmma_f32_16x16x32_bf16(false, at.v, false, b.v, (short)0, ct1, false, false);
                ca1 = __builtin_amdgcn_wmma_f32_16x16x32_bf16(false, aa.v, false, b.v, (short)0, ca1, false, false);
                break;
            case 2:
                ct2 = __builtin_amdgcn_wmma_f32_16x16x32_bf16(false, at.v, false, b.v, (short)0, ct2, false, false);
                ca2 = __builtin_amdgcn_wmma_f32_16x16x32_bf16(false, aa.v, false, b.v, (short)0, ca2, false, false);
                break;
            default:
                ct3 = __builtin_amdgcn_wmma_f32_16x16x32_bf16(false, at.v, false, b.v, (short)0, ct3, false, false);
                ca3 = __builtin_amdgcn_wmma_f32_16x16x32_bf16(false, aa.v, false, b.v, (short)0, ca3, false, false);
                break;
            }
        }

        if (more) {
            const int nb = buf ^ 1;
            *(u32x4*)(&sAt[nb][ar * LPITCH + ac])  = rt;
            *(u32x4*)(&sAa[nb][ar * LPITCH + ac])  = rq;
            *(u32x4*)(&sB[nb][bn0 * LPITCH + bc])  = rb0;
            *(u32x4*)(&sB[nb][bn1 * LPITCH + bc])  = rb1;
            __syncthreads();
            buf = nb;
        }
    }

    const int m0 = m0b + wm * 16;
    const int n0 = n0b + wn * 64;
    v8f rec[4] = {ct0, ct1, ct2, ct3};
    v8f aux[4] = {ca0, ca1, ca2, ca3};
    float l2s = 0.0f, axs = 0.0f;
    #pragma unroll
    for (int t = 0; t < 4; ++t) {
        const int col = n0 + t * 16 + l15;
        const float bd = b_dec[col];
        #pragma unroll
        for (int r = 0; r < 8; ++r) {
            const int m = m0 + r + half * 8;
            const float xr = rec[t][r] + bd;
            out[(size_t)m * N + col] = xr;
            const float xv = x[(size_t)m * N + col];
            const float d  = xr - xv;
            l2s += d * d;
            const float ad = (xv - xr) - aux[t][r];  // residual - x_aux
            axs += ad * ad;
        }
    }

    __shared__ float s1[256], s2[256];
    s1[threadIdx.x] = l2s;
    s2[threadIdx.x] = axs;
    __syncthreads();
    for (int s = 128; s > 0; s >>= 1) {
        if (threadIdx.x < s) {
            s1[threadIdx.x] += s1[threadIdx.x + s];
            s2[threadIdx.x] += s2[threadIdx.x + s];
        }
        __syncthreads();
    }
    if (threadIdx.x == 0) {
        const int bid = blockIdx.y * gridDim.x + blockIdx.x;
        partials[2 * bid + 0] = s1[0];
        partials[2 * bid + 1] = s2[0];
    }
}

// ------------------------------- finalize ----------------------------------

__global__ __launch_bounds__(256) void finalize_kernel(const float* __restrict__ partials,
                                                       int nparts,
                                                       const unsigned int* __restrict__ selcount,
                                                       const int* __restrict__ nba,
                                                       float* __restrict__ scalars) {
    __shared__ float s1[256], s2[256];
    __shared__ int sd[256];
    float l2 = 0.0f, ax = 0.0f;
    int dead = 0;
    for (int i = threadIdx.x; i < nparts; i += 256) {
        l2 += partials[2 * i + 0];
        ax += partials[2 * i + 1];
    }
    for (int j = threadIdx.x; j < HID_DIM; j += 256) dead |= (nba[j] >= N_DEAD) ? 1 : 0;
    s1[threadIdx.x] = l2;
    s2[threadIdx.x] = ax;
    sd[threadIdx.x] = dead;
    __syncthreads();
    for (int s = 128; s > 0; s >>= 1) {
        if (threadIdx.x < s) {
            s1[threadIdx.x] += s1[threadIdx.x + s];
            s2[threadIdx.x] += s2[threadIdx.x + s];
            sd[threadIdx.x] |= sd[threadIdx.x + s];
        }
        __syncthreads();
    }
    if (threadIdx.x == 0) {
        const float denom = (float)BATCH * (float)IN_DIM;
        const float l2_loss  = s1[0] / denom;
        const float aux_loss = sd[0] ? (AUX_PEN * s2[0] / denom) : 0.0f;
        const float l0_norm  = (float)(*selcount) / (float)BATCH;
        scalars[0] = l2_loss + aux_loss;  // loss
        scalars[1] = l2_loss;             // l2_loss
        scalars[2] = l0_norm;             // l0_norm
        scalars[3] = aux_loss;            // aux_loss
    }
}

// ------------------------------- launcher ----------------------------------

extern "C" void kernel_launch(void* const* d_in, const int* in_sizes, int n_in,
                              void* d_out, int out_size, void* d_ws, size_t ws_size,
                              hipStream_t stream) {
    const float* x     = (const float*)d_in[0];
    const float* W_enc = (const float*)d_in[1];
    const float* W_dec = (const float*)d_in[2];
    const float* b_dec = (const float*)d_in[3];
    const int*   nba   = (const int*)d_in[4];
    float* out = (float*)d_out;

    char* ws = (char*)d_ws;
    size_t off = 0;
    auto alloc = [&](size_t bytes) { char* p = ws + off; off = (off + bytes + 255) & ~(size_t)255; return p; };

    __bf16* xc    = (__bf16*)alloc((size_t)BATCH * IN_DIM * 2);
    __bf16* WencT = (__bf16*)alloc((size_t)HID_DIM * IN_DIM * 2);
    __bf16* WdecT = (__bf16*)alloc((size_t)IN_DIM * HID_DIM * 2);
    float*  pre   = (float*)alloc((size_t)BATCH * HID_DIM * 4);
    __bf16* Atopk = (__bf16*)alloc((size_t)BATCH * HID_DIM * 2);
    __bf16* Aaux  = (__bf16*)alloc((size_t)BATCH * HID_DIM * 2);
    unsigned int* hist1 = (unsigned int*)alloc(NBINS * 4);
    unsigned int* hist2 = (unsigned int*)alloc(NBINS * 4);
    unsigned int* ctrl  = (unsigned int*)alloc(256);      // [0]=t1 [1]=rem [2]=thr [3]=selcount
    const int decBlocks = (IN_DIM / 128) * (BATCH / 64);  // 8 * 32 = 256
    float* partials = (float*)alloc((size_t)decBlocks * 2 * 4);

    // zero histograms / control (ws is NOT re-poisoned between replays)
    hipMemsetAsync(hist1, 0, NBINS * 4, stream);
    hipMemsetAsync(hist2, 0, NBINS * 4, stream);
    hipMemsetAsync(ctrl, 0, 256, stream);

    // 1) centered input -> bf16
    prep_x_kernel<<<2048, 256, 0, stream>>>(x, b_dec, xc);

    // 2) weight transpose-converts (K-major bf16 for WMMA B-fragments)
    transpose_convert_kernel<<<dim3(HID_DIM / 32, IN_DIM / 32), 256, 0, stream>>>(W_enc, WencT, IN_DIM, HID_DIM);
    transpose_convert_kernel<<<dim3(IN_DIM / 32, HID_DIM / 32), 256, 0, stream>>>(W_dec, WdecT, HID_DIM, IN_DIM);

    // 3) encoder GEMM -> pre_acts
    gemm_enc_kernel<<<dim3(HID_DIM / 128, BATCH / 64), 256, 0, stream>>>(xc, WencT, pre);

    // 4) global top-k threshold via two-level radix select
    hist1_kernel<<<2048, 256, 0, stream>>>(pre, hist1);
    scan1_kernel<<<1, 32, 0, stream>>>(hist1, ctrl);
    hist2_kernel<<<2048, 256, 0, stream>>>(pre, ctrl, hist2);
    scan2_kernel<<<1, 32, 0, stream>>>(hist2, ctrl);

    // 5) scatter top-k into bf16 A matrix (+ exact l0 count)
    scatter_topk_kernel<<<2048, 256, 0, stream>>>(pre, ctrl, Atopk, &ctrl[3]);

    // 6) per-row aux top-64 over dead features
    aux_topk_kernel<<<BATCH, 256, 0, stream>>>(pre, nba, Aaux);

    // 7) fused decoder GEMM (x_rec + x_aux) with loss partials
    gemm_dec_kernel<<<dim3(IN_DIM / 128, BATCH / 64), 256, 0, stream>>>(
        Atopk, Aaux, WdecT, x, b_dec, out, partials);

    // 8) final scalar reduction
    finalize_kernel<<<1, 256, 0, stream>>>(partials, decBlocks, &ctrl[3], nba,
                                           out + (size_t)BATCH * IN_DIM);
}
